// STAN_74783970558565
// MI455X (gfx1250) — compile-verified, matrix operand order checked by reading, not run on hardware
//
#include <hip/hip_runtime.h>
#include <hip/hip_bf16.h>
#include <math.h>

typedef __attribute__((ext_vector_type(16))) __bf16 v16bf;
typedef __attribute__((ext_vector_type(8)))  __bf16 v8bf;
typedef __attribute__((ext_vector_type(8)))  float  v8f;

#define N_  32
#define M_  128
#define L_  128
#define E_  128
#define HOURS_ 24

// ---- WMMA fragment helpers (16x16x32 bf16, wave32 layouts per CDNA5 ISA) ----

// A/B fragment from a row-major [16 x 128] tile in LDS.
// Lane holds row (lane%16); elements 0..7 -> k = kbase + 8*(lane/16) + 0..7,
// elements 8..15 -> k = kbase + 16 + 8*(lane/16) + 0..7 (ISA 16-bit A 16x32 layout).
__device__ __forceinline__ v16bf frag_rowmajor_lds(const __bf16* base, int kbase) {
  int lane = threadIdx.x & 31;
  int row  = lane & 15;
  int koff = (lane >> 4) << 3;
  const __bf16* p = base + row * E_ + kbase + koff;
  v8bf lo = *(const v8bf*)(p);
  v8bf hi = *(const v8bf*)(p + 16);
  return __builtin_shufflevector(lo, hi, 0,1,2,3,4,5,6,7,8,9,10,11,12,13,14,15);
}

// Same pattern, but source is row-major f32 in global memory (Wq/Wk/Wv rows).
__device__ __forceinline__ v16bf frag_rowmajor_f32(const float* base, int kbase) {
  int lane = threadIdx.x & 31;
  int row  = lane & 15;
  int koff = (lane >> 4) << 3;
  const float* p = base + row * E_ + kbase + koff;
  v16bf r;
#pragma unroll
  for (int i = 0; i < 8; ++i) r[i] = (__bf16)p[i];
#pragma unroll
  for (int i = 0; i < 8; ++i) r[i + 8] = (__bf16)p[16 + i];
  return r;
}

// B fragment where logical B[k][n] lives at base[k*128 + n] (e.g. V[m][e] with k=m).
__device__ __forceinline__ v16bf frag_colmajor_lds(const __bf16* base, int kbase) {
  int lane = threadIdx.x & 31;
  int col  = lane & 15;
  int koff = (lane >> 4) << 3;
  v16bf r;
#pragma unroll
  for (int i = 0; i < 8; ++i) r[i]     = base[(kbase + koff + i)      * E_ + col];
#pragma unroll
  for (int i = 0; i < 8; ++i) r[i + 8] = base[(kbase + 16 + koff + i) * E_ + col];
  return r;
}

// Store a 16x16 f32 accumulator tile into a row-major bf16 LDS tile.
// C layout: lane -> col (lane%16); VGPR r -> row r + 8*(lane/16).
__device__ __forceinline__ void store_tile(__bf16* dst, const v8f c) {
  int lane  = threadIdx.x & 31;
  int col   = lane & 15;
  int rbase = (lane >> 4) << 3;
#pragma unroll
  for (int r = 0; r < 8; ++r) dst[(rbase + r) * E_ + col] = (__bf16)c[r];
}

__device__ __forceinline__ v8f wmma_bf16(v16bf a, v16bf b, v8f c) {
  return __builtin_amdgcn_wmma_f32_16x16x32_bf16(false, a, false, b, (short)0, c, false, false);
}

__global__ __launch_bounds__(256) void stan_fused_kernel(
    const int* __restrict__ full_seq, const int* __restrict__ time_seq,
    const int* __restrict__ user, const int* __restrict__ posneg,
    const int* __restrict__ traj_len,
    const float* __restrict__ mat1, const float* __restrict__ mat2,
    const float* __restrict__ vec,
    const float* __restrict__ emb_t, const float* __restrict__ emb_u,
    const float* __restrict__ emb_loc,
    const float* __restrict__ emb_sl, const float* __restrict__ emb_su,
    const float* __restrict__ emb_tl, const float* __restrict__ emb_tu,
    const float* __restrict__ Wq, const float* __restrict__ Wk,
    const float* __restrict__ Wv, float* __restrict__ out)
{
  // Buffer reuse over phases: buf0 = joint -> self_attn, buf1 = Q -> cand,
  // buf2 = K -> attn probs P, buf3 = V.   4 x 32KB = 128KB LDS (<320KB/WGP).
  __shared__ __align__(16) __bf16 buf0[M_ * E_];
  __shared__ __align__(16) __bf16 buf1[M_ * E_];
  __shared__ __align__(16) __bf16 buf2[M_ * E_];
  __shared__ __align__(16) __bf16 buf3[M_ * E_];
  __shared__ float ssum[8]; // [sl0,sl1,su0,su1,tl0,tl1,tu0,tu1] row-sums over E

  const int b    = blockIdx.x;
  const int tid  = threadIdx.x;
  const int lane = tid & 31;
  const int w    = tid >> 5;      // wave id 0..7
  const int cn   = lane & 15;     // column within 16x16 tile
  const int half = lane >> 4;     // 0 or 1 (row offset 0/8 in C layout)
  const int tl   = traj_len[b];
  const int u    = user[b];

  // ---- Phase 0: the 8 scalar table row-sums (delta collapses to these) ----
  {
    const float* tab = (w < 2) ? emb_sl : (w < 4) ? emb_su : (w < 6) ? emb_tl : emb_tu;
    const float* rowp = tab + (w & 1) * E_;
    float p = rowp[lane] + rowp[lane + 32] + rowp[lane + 64] + rowp[lane + 96];
#pragma unroll
    for (int o = 16; o >= 1; o >>= 1) p += __shfl_xor(p, o, 32);
    if (lane == 0) ssum[w] = p;
  }

  // ---- Phase A: joint = emb_t[tim] + emb_loc[full_seq] + emb_u[user] (bf16) --
  for (int idx = tid; idx < M_ * E_; idx += 256) {
    int m = idx >> 7, e = idx & 127;
    int ts  = time_seq[b * M_ + m];
    int tim = (ts - 1) % HOURS_ + 1;
    int fs  = full_seq[b * M_ + m];
    float v = emb_t[tim * E_ + e] + emb_loc[(size_t)fs * E_ + e]
            + emb_u[(size_t)u * E_ + e];
    buf0[idx] = (__bf16)v;
  }
  __syncthreads(); // B1

  // ---- Phase B: Q/K/V = joint @ Wᵀ, 192 16x16 tiles over 8 waves ----------
  for (int t = w; t < 192; t += 8) {
    int which = t >> 6;
    int rem   = t & 63;
    int ti = rem >> 3, tn = rem & 7;
    const float* Wp = (which == 0) ? Wq : (which == 1) ? Wk : Wv;
    __bf16* Op = (which == 0) ? buf1 : (which == 1) ? buf2 : buf3;
    v8f acc = {};
#pragma unroll
    for (int kk = 0; kk < 4; ++kk) {
      v16bf a  = frag_rowmajor_lds(buf0 + ti * 16 * E_, kk * 32);
      v16bf bb = frag_rowmajor_f32(Wp + tn * 16 * E_, kk * 32);
      acc = wmma_bf16(a, bb, acc);
    }
    store_tile(Op + ti * 16 * E_ + tn * 16, acc);
  }
  __syncthreads(); // B2

  // ---- Phase C: S = QKᵀ + delta ; row softmax ; fmask -> P (regs) ---------
  const int rowbase = w * 16;
  v8f S[8];
  {
    v16bf aQ[4];
#pragma unroll
    for (int kk = 0; kk < 4; ++kk)
      aQ[kk] = frag_rowmajor_lds(buf1 + rowbase * E_, kk * 32);
#pragma unroll
    for (int t = 0; t < 8; ++t) {
      v8f acc = {};
#pragma unroll
      for (int kk = 0; kk < 4; ++kk)
        acc = wmma_bf16(aQ[kk], frag_rowmajor_lds(buf2 + t * 16 * E_, kk * 32), acc);
      S[t] = acc;
    }
  }
  {
    float s0 = ssum[0], s1 = ssum[1], s2 = ssum[2], s3 = ssum[3];
    float s4 = ssum[4], s5 = ssum[5], s6 = ssum[6], s7 = ssum[7];
#pragma unroll
    for (int t = 0; t < 8; ++t) {
      int j  = t * 16 + cn;
      int vj = (j < tl) ? 1 : 0;
#pragma unroll
      for (int r = 0; r < 8; ++r) {
        int i = rowbase + r + 8 * half;
        bool msk = (i < tl) && vj;
        const float* dp = mat1 + (((size_t)b * M_ + i) * M_ + j) * 2;
        float ds = dp[0], dt = dp[1];
        float sl = msk ? s1 : s0, su = msk ? s3 : s2;
        float tlo = msk ? s5 : s4, tu = msk ? s7 : s6;
        S[t][r] += (sl * (100.f - ds) + su * ds + tlo * (100.f - dt) + tu * dt) * 0.01f;
      }
    }
  }
  // Row-wise softmax: 8 col-tiles in-register + 16 lanes of own half.
#pragma unroll
  for (int r = 0; r < 8; ++r) {
    float mx = S[0][r];
#pragma unroll
    for (int t = 1; t < 8; ++t) mx = fmaxf(mx, S[t][r]);
#pragma unroll
    for (int o = 1; o < 16; o <<= 1) mx = fmaxf(mx, __shfl_xor(mx, o, 32));
    float sm = 0.f;
#pragma unroll
    for (int t = 0; t < 8; ++t) { float e = expf(S[t][r] - mx); S[t][r] = e; sm += e; }
#pragma unroll
    for (int o = 1; o < 16; o <<= 1) sm += __shfl_xor(sm, o, 32);
    float inv = 1.f / sm;
#pragma unroll
    for (int t = 0; t < 8; ++t) S[t][r] *= inv;
  }
  __syncthreads(); // B3: all waves done reading Q(buf1)/K(buf2)

  // ---- Phase D: write P (fmask applied) into buf2; gather cand into buf1 --
#pragma unroll
  for (int t = 0; t < 8; ++t) {
    int j  = t * 16 + cn;
    int vj = (j < tl);
#pragma unroll
    for (int r = 0; r < 8; ++r) {
      int i = rowbase + r + 8 * half;
      float p = ((i < tl) && vj) ? S[t][r] : 0.f;
      buf2[i * E_ + j] = (__bf16)p;
    }
  }
  for (int idx = tid; idx < L_ * E_; idx += 256) {
    int l = idx >> 7, e = idx & 127;
    int pn = posneg[b * L_ + l];
    buf1[idx] = (__bf16)emb_loc[(size_t)pn * E_ + e];
  }

  // ---- Phase E: self_attn = P @ V (own 16 rows; same-wave LDS is in-order) -
  {
    v16bf aP[4];
#pragma unroll
    for (int kk = 0; kk < 4; ++kk)
      aP[kk] = frag_rowmajor_lds(buf2 + rowbase * E_, kk * 32);
#pragma unroll
    for (int t = 0; t < 8; ++t) {
      v8f acc = {};
#pragma unroll
      for (int kk = 0; kk < 4; ++kk)
        acc = wmma_bf16(aP[kk], frag_colmajor_lds(buf3 + t * 16, kk * 32), acc);
      store_tile(buf0 + rowbase * E_ + t * 16, acc);
    }
  }
  __syncthreads(); // B4: self_attn (buf0) + cand (buf1) complete

  // ---- Phase F: G = cand @ self_attnᵀ, weight by self_delta, reduce over m -
  {
    float s0 = ssum[0], s1 = ssum[1], s2 = ssum[2], s3 = ssum[3];
    float s4 = ssum[4], s5 = ssum[5], s6 = ssum[6], s7 = ssum[7];
    v16bf aC[4];
#pragma unroll
    for (int kk = 0; kk < 4; ++kk)
      aC[kk] = frag_rowmajor_lds(buf1 + rowbase * E_, kk * 32);
    float partial[8];
#pragma unroll
    for (int r = 0; r < 8; ++r) partial[r] = 0.f;
#pragma unroll
    for (int t = 0; t < 8; ++t) {
      v8f acc = {};
#pragma unroll
      for (int kk = 0; kk < 4; ++kk)
        acc = wmma_bf16(aC[kk], frag_rowmajor_lds(buf0 + t * 16 * E_, kk * 32), acc);
      int  m  = t * 16 + cn;
      bool vm = (m < tl);
      float sl = vm ? s1 : s0, su = vm ? s3 : s2;
      float tlo = vm ? s5 : s4, tu = vm ? s7 : s6;
      float vecm = vec[b * M_ + m];
      float c1 = (su - sl) * 0.01f;
      float c0 = sl + (tlo * (100.f - vecm) + tu * vecm) * 0.01f;
      const float* m2 = mat2 + ((size_t)b * M_ + m) * L_;
#pragma unroll
      for (int r = 0; r < 8; ++r) {
        int l = rowbase + r + 8 * half;
        partial[r] += acc[r] * (c0 + c1 * m2[l]);
      }
    }
#pragma unroll
    for (int r = 0; r < 8; ++r) {
      float p = partial[r];
#pragma unroll
      for (int o = 1; o < 16; o <<= 1) p += __shfl_xor(p, o, 32);
      if (cn == 0) out[b * L_ + rowbase + r + 8 * half] = p;
    }
  }
}

extern "C" void kernel_launch(void* const* d_in, const int* in_sizes, int n_in,
                              void* d_out, int out_size, void* d_ws, size_t ws_size,
                              hipStream_t stream) {
  (void)in_sizes; (void)n_in; (void)out_size; (void)d_ws; (void)ws_size;
  stan_fused_kernel<<<N_, 256, 0, stream>>>(
      (const int*)d_in[0],   (const int*)d_in[1],   (const int*)d_in[2],
      (const int*)d_in[3],   (const int*)d_in[4],
      (const float*)d_in[5], (const float*)d_in[6], (const float*)d_in[7],
      (const float*)d_in[8], (const float*)d_in[9], (const float*)d_in[10],
      (const float*)d_in[11],(const float*)d_in[12],(const float*)d_in[13],
      (const float*)d_in[14],(const float*)d_in[15],(const float*)d_in[16],
      (const float*)d_in[17],(float*)d_out);
}
